// GCN_Critic_39101382262872
// MI455X (gfx1250) — matrix operand order repaired and checked by reading, not compiled, not run on hardware
//
#include <hip/hip_runtime.h>
#include <hip/hip_bf16.h>

typedef __attribute__((ext_vector_type(16))) _Float16 v16h;
typedef __attribute__((ext_vector_type(8)))  float    v8f;
typedef __attribute__((ext_vector_type(4)))  unsigned int u32x4;
typedef __attribute__((ext_vector_type(8)))  int      i32x8;
typedef __attribute__((ext_vector_type(4)))  int      i32x4;

namespace {

constexpr int B  = 64;
constexpr int N  = 512;
constexpr int E  = 16384;
constexpr int F  = 128;          // HID == EMB == 128
constexpr int BN = B * N;        // 32768 nodes total

union Frag { v16h v; uint4 q[2]; };
union H4   { _Float16 h[4]; uint2 u; };

// ---------------------------------------------------------------- utilities
__global__ void k_zero_u32(unsigned* __restrict__ p, unsigned n) {
  unsigned t = blockIdx.x * blockDim.x + threadIdx.x;
  if (t < n) p[t] = 0u;
}

// Convert the three 128x128 f32 weight matrices to f16, TRANSPOSED (Wt[n][k])
// so WMMA B-fragments become two contiguous 8-half chunks per lane.
__global__ void k_cvt_w(const float* __restrict__ W2, const float* __restrict__ M1,
                        const float* __restrict__ M2, _Float16* __restrict__ W2t,
                        _Float16* __restrict__ M1t, _Float16* __restrict__ M2t) {
  unsigned t = blockIdx.x * blockDim.x + threadIdx.x;
  if (t >= 3u * F * F) return;
  unsigned m = t / (F * F), idx = t % (F * F);
  unsigned k = idx >> 7, n = idx & 127u;          // in[k][n]
  if      (m == 0) W2t[n * F + k] = (_Float16)W2[idx];
  else if (m == 1) M1t[n * F + k] = (_Float16)M1[idx];
  else             M2t[n * F + k] = (_Float16)M2[idx];
}

// ---------------------------------------------------------------- degrees
__global__ void k_deg(const int* __restrict__ ei, unsigned* __restrict__ deg) {
  unsigned t = blockIdx.x * blockDim.x + threadIdx.x;   // one thread per edge
  if (t >= (unsigned)(B * E)) return;
  unsigned b = t >> 14, e = t & (unsigned)(E - 1);      // E == 2^14
  int d = ei[b * 2 * E + E + e];                        // edge_index[b][1][e]
  atomicAdd(&deg[b * N + d], 1u);
}

__global__ void k_dinv(const unsigned* __restrict__ deg, float* __restrict__ dinv) {
  unsigned t = blockIdx.x * blockDim.x + threadIdx.x;
  if (t < (unsigned)BN) dinv[t] = rsqrtf((float)deg[t] + 1.0f);
}

// ------------------------------------------------- conv1: x(3) @ W1(3x128), f16 out
__global__ void k_conv1(const int* __restrict__ actions, const float* __restrict__ emb,
                        const float* __restrict__ nf, const float* __restrict__ W1,
                        _Float16* __restrict__ xh) {
  unsigned t = blockIdx.x * blockDim.x + threadIdx.x;   // BN*32 items, 4 feats each
  if (t >= (unsigned)BN * 32u) return;
  unsigned g = t >> 5, f4 = (t & 31u) * 4u;
  unsigned b = g >> 9, n = g & 511u;                    // N == 2^9
  int ia = actions[2 * b], id = actions[2 * b + 1];
  float xa = emb[(size_t)ia * N + n];
  float xd = emb[(size_t)id * N + n];
  float xn = nf[g];
  const float4 w0 = *(const float4*)(W1 + f4);          // W1[0][f4..]
  const float4 w1 = *(const float4*)(W1 + F + f4);      // W1[1][f4..]
  const float4 w2 = *(const float4*)(W1 + 2 * F + f4);  // W1[2][f4..]
  H4 o;
  o.h[0] = (_Float16)(xa * w0.x + xd * w1.x + xn * w2.x);
  o.h[1] = (_Float16)(xa * w0.y + xd * w1.y + xn * w2.y);
  o.h[2] = (_Float16)(xa * w0.z + xd * w1.z + xn * w2.z);
  o.h[3] = (_Float16)(xa * w0.w + xd * w1.w + xn * w2.w);
  *(uint2*)(xh + (size_t)g * F + f4) = o.u;
}

// ------------------------------------------------- edge scatter: agg[dst] += h[src]*norm
__global__ void k_scatter(const int* __restrict__ ei, const float* __restrict__ dinv,
                          const _Float16* __restrict__ h, float* __restrict__ agg) {
  unsigned t = blockIdx.x * blockDim.x + threadIdx.x;   // B*E*32 items (4 feats each)
  if (t >= (unsigned)(B * E) * 32u) return;
  unsigned fi   = (t & 31u) * 4u;
  unsigned edge = t >> 5;
  unsigned b = edge >> 14, e = edge & (unsigned)(E - 1);
  int s = ei[b * 2 * E + e];
  int d = ei[b * 2 * E + E + e];
  unsigned gs = b * N + s, gd = b * N + d;
  float norm = dinv[gs] * dinv[gd];
  H4 v; v.u = *(const uint2*)(h + (size_t)gs * F + fi);
  float* ap = agg + (size_t)gd * F + fi;
  atomicAdd(ap + 0, (float)v.h[0] * norm);
  atomicAdd(ap + 1, (float)v.h[1] * norm);
  atomicAdd(ap + 2, (float)v.h[2] * norm);
  atomicAdd(ap + 3, (float)v.h[3] * norm);
}

// ------------------------------------------------- finalize: agg + h*dinv^2 + bias (+relu)
__global__ void k_finalize(const float* __restrict__ agg_in, const _Float16* __restrict__ hin,
                           const float* __restrict__ dinv, const float* __restrict__ bias,
                           _Float16* __restrict__ xout, float* __restrict__ agg_clear,
                           int relu) {
  unsigned t = blockIdx.x * blockDim.x + threadIdx.x;   // BN*F
  if (t >= (unsigned)BN * (unsigned)F) return;
  unsigned g = t >> 7, f = t & 127u;
  float di = dinv[g];
  float v = agg_in[t] + (float)hin[t] * (di * di) + bias[f];
  if (relu) v = fmaxf(v, 0.0f);
  xout[t] = (_Float16)v;
  if (agg_clear) agg_clear[t] = 0.0f;                   // re-zero for next scatter
}

// ------------------------------------------------- WMMA GEMM: Y(Mx128) = X(Mx128) @ W(128x128)
// Wt (W transposed: Wt[n][k]) is DMA'd into LDS by the Tensor Data Mover.
// One wave -> 16-row x 128-col strip; A fragments preloaded and reused across
// all 8 column tiles; B fragments batch-loaded (one dscnt wait per 4 WMMAs).
__global__ __launch_bounds__(256) void k_gemm(const _Float16* __restrict__ X,
                                              const _Float16* __restrict__ Wt,
                                              const float* __restrict__ bias,
                                              _Float16* __restrict__ Y, int relu) {
  __shared__ __align__(16) _Float16 lw[F * F];          // 32 KB

#if defined(__gfx1250__)
  // --- TDM stage: wave 0 DMAs the whole 32 KB weight tile into LDS -------
  if (threadIdx.x < 32) {
    unsigned long long ga = (unsigned long long)(uintptr_t)Wt;
    unsigned lds_off = (unsigned)(uintptr_t)lw;        // low 32 bits = LDS byte offset
    // D# group 0: count=1 (valid), lds_addr, global_addr=Wt, type=2
    u32x4 g0;
    g0[0] = 1u;                                        // count=1, no gather
    g0[1] = lds_off;                                   // LDS destination offset
    g0[2] = (unsigned)(ga & 0xffffffffu);              // global_addr[31:0]
    g0[3] = (unsigned)((ga >> 32) & 0x01ffffffu) | (2u << 30);  // addr[56:32], type=2
    // D# group 1: 1-D tensor of 16384 2-byte elements, tile == tensor
    i32x8 g1;
    g1[0] = 0x00010000;          // workgroup_mask=0, data_size=1 (2B), no flags
    g1[1] = 0x40000000;          // tensor_dim0 low16 (=0x4000) in bits 63:48
    g1[2] = 0x00010000;          // tensor_dim0 hi16=0 ; tensor_dim1 low16=1
    g1[3] = 0x40000000;          // tensor_dim1 hi16=0 ; tile_dim0=0x4000
    g1[4] = 0x00000001;          // tile_dim1=1 ; tile_dim2=0
    g1[5] = 16384;               // tensor_dim0_stride[31:0]
    g1[6] = 0x40000000;          // stride0[47:32]=0 ; tensor_dim1_stride low16=0x4000
    g1[7] = 0;                   // tensor_dim1_stride high bits
    i32x4 gz4 = {0, 0, 0, 0};    // groups 2/3 unused (tile_dim2..4 = 0)
    i32x8 gz8 = {0, 0, 0, 0, 0, 0, 0, 0};
    __builtin_amdgcn_tensor_load_to_lds(g0, g1, gz4, gz4, gz8, 0);
    __builtin_amdgcn_s_wait_tensorcnt(0);
  }
  __syncthreads();
  // The TDM builtin carries no pointer operand, so clang cannot see that it
  // writes `lw`. This opaque may-write barrier prevents the B-fragment LDS
  // loads below from being folded to undef (round-3 regression: ds=0).
  asm volatile("" ::: "memory");
#else
  // Host-pass / non-gfx1250 fallback: cooperative vector copy.
  for (int i = threadIdx.x; i < (F * F) / 8; i += 256)
    ((uint4*)lw)[i] = ((const uint4*)Wt)[i];
  __syncthreads();
#endif

  const int t = threadIdx.x;
  const int wave = t >> 5, lane = t & 31;
  const int m0 = (blockIdx.x * 8 + wave) * 16;          // grid sized exactly: BN/128
  const int lm = lane & 15, lh = lane >> 4;
  const _Float16* xrow = X + (size_t)(m0 + lm) * F;
  __builtin_prefetch(xrow, 0, 0);                       // global_prefetch_b8

  // A fragments for all 4 K-steps (16x32 f16 each):
  // lane holds K = kb+lh*8 .. +7 and kb+16+lh*8 .. +7 (two contiguous 16B chunks)
  Frag a[4];
#pragma unroll
  for (int kk = 0; kk < 4; ++kk) {
    const _Float16* ap = xrow + kk * 32 + lh * 8;
    a[kk].q[0] = *(const uint4*)ap;
    a[kk].q[1] = *(const uint4*)(ap + 16);
  }

#pragma unroll
  for (int nt = 0; nt < 8; ++nt) {
    const int col = nt * 16 + lm;
    // B fragments (32x16 f16): lane (col) holds K = kb+lh*8 .. +7 and kb+16+lh*8 .. +7 in Wt row
    const _Float16* bcol = lw + col * F;
    Frag bf[4];
#pragma unroll
    for (int kk = 0; kk < 4; ++kk) {
      const _Float16* bp = bcol + kk * 32 + lh * 8;
      bf[kk].q[0] = *(const uint4*)bp;
      bf[kk].q[1] = *(const uint4*)(bp + 16);
    }
    v8f acc = {};
#pragma unroll
    for (int kk = 0; kk < 4; ++kk)
      acc = __builtin_amdgcn_wmma_f32_16x16x32_f16(
          /*neg_a=*/false, a[kk].v, /*neg_b=*/false, bf[kk].v,
          /*c_mod=*/(short)0, acc, /*reuse_a=*/false, /*reuse_b=*/false);

    // D layout: VGPR r -> row m0 + lh*8 + r, col nt*16 + lm
    const float bv = bias ? bias[col] : 0.0f;
#pragma unroll
    for (int r = 0; r < 8; ++r) {
      float v = acc[r] + bv;
      if (relu) v = fmaxf(v, 0.0f);
      Y[(size_t)(m0 + lh * 8 + r) * F + col] = (_Float16)v;
    }
  }
}

// ------------------------------------------------- output head: sigmoid(x . outW + b)
__global__ void k_out(const _Float16* __restrict__ x, const float* __restrict__ outW,
                      const float* __restrict__ outB, float* __restrict__ out) {
  unsigned g = blockIdx.x * blockDim.x + threadIdx.x;
  if (g >= (unsigned)BN) return;
  float s = outB[0];
  const _Float16* xr = x + (size_t)g * F;
#pragma unroll 4
  for (int f = 0; f < F; ++f) s += (float)xr[f] * outW[f];
  out[g] = 1.0f / (1.0f + __expf(-s));
}

}  // namespace

extern "C" void kernel_launch(void* const* d_in, const int* in_sizes, int n_in,
                              void* d_out, int out_size, void* d_ws, size_t ws_size,
                              hipStream_t stream) {
  (void)in_sizes; (void)n_in; (void)out_size; (void)ws_size;

  const int*   actions = (const int*)  d_in[0];
  const float* nodef   = (const float*)d_in[1];
  const int*   ei      = (const int*)  d_in[2];
  const float* emb     = (const float*)d_in[3];
  const float* convW1  = (const float*)d_in[4];
  const float* convB1  = (const float*)d_in[5];
  const float* convW2  = (const float*)d_in[6];
  const float* convB2  = (const float*)d_in[7];
  const float* mlpW1   = (const float*)d_in[8];
  const float* mlpB1   = (const float*)d_in[9];
  const float* mlpW2   = (const float*)d_in[10];
  const float* mlpB2   = (const float*)d_in[11];
  const float* outW    = (const float*)d_in[12];
  const float* outB    = (const float*)d_in[13];
  float* out = (float*)d_out;

  // -------- workspace layout (~32.4 MB), all sub-buffers 16B aligned --------
  char* ws = (char*)d_ws;
  _Float16* xh  = (_Float16*)ws;                 // BN*F f16   (8 MB)
  _Float16* yh  = xh + (size_t)BN * F;           // BN*F f16   (8 MB)
  float*    agg = (float*)(yh + (size_t)BN * F); // BN*F f32   (16 MB)
  unsigned* deg = (unsigned*)(agg + (size_t)BN * F);   // BN u32
  float*    din = (float*)(deg + BN);                  // BN f32
  _Float16* W2t = (_Float16*)(din + BN);               // 128x128 f16
  _Float16* M1t = W2t + F * F;
  _Float16* M2t = M1t + F * F;

  const int TB = 256;

  // 1. zero accumulators + degree counters
  k_zero_u32<<<(BN * F + TB - 1) / TB, TB, 0, stream>>>((unsigned*)agg, BN * F);
  k_zero_u32<<<(BN + TB - 1) / TB, TB, 0, stream>>>(deg, BN);

  // 2. f16 transposed weights
  k_cvt_w<<<(3 * F * F + TB - 1) / TB, TB, 0, stream>>>(convW2, mlpW1, mlpW2, W2t, M1t, M2t);

  // 3. degrees + rsqrt
  k_deg<<<(B * E + TB - 1) / TB, TB, 0, stream>>>(ei, deg);
  k_dinv<<<(BN + TB - 1) / TB, TB, 0, stream>>>(deg, din);

  // 4. conv1 dense part (3->128), f16 activations
  k_conv1<<<(BN * 32 + TB - 1) / TB, TB, 0, stream>>>(actions, emb, nodef, convW1, xh);

  // 5. conv1 scatter + finalize (ReLU, bias1, re-zero agg)
  k_scatter<<<(B * E * 32) / TB, TB, 0, stream>>>(ei, din, xh, agg);
  k_finalize<<<(BN * F) / TB, TB, 0, stream>>>(agg, xh, din, convB1, xh, agg, 1);

  // 6. conv2 dense part via WMMA (no bias/relu here; added in finalize2)
  k_gemm<<<BN / 128, TB, 0, stream>>>(xh, W2t, nullptr, yh, 0);

  // 7. conv2 scatter + finalize (no ReLU, bias2)
  k_scatter<<<(B * E * 32) / TB, TB, 0, stream>>>(ei, din, yh, agg);
  k_finalize<<<(BN * F) / TB, TB, 0, stream>>>(agg, yh, din, convB2, xh, nullptr, 0);

  // 8. MLP: two WMMA GEMMs with fused bias+ReLU
  k_gemm<<<BN / 128, TB, 0, stream>>>(xh, M1t, mlpB1, yh, 1);
  k_gemm<<<BN / 128, TB, 0, stream>>>(yh, M2t, mlpB2, xh, 1);

  // 9. output head + sigmoid
  k_out<<<(BN + TB - 1) / TB, TB, 0, stream>>>(xh, outW, outB, out);
}
